// ParticleAttentionBlock_81664508166584
// MI455X (gfx1250) — matrix operand order repaired and checked by reading, not compiled, not run on hardware
//
#include <hip/hip_runtime.h>

// Problem geometry (fixed by the reference)
#define B_    16
#define N_    512
#define D_    512
#define H_    8
#define HD_   64
#define UE_   6
#define DFF_  2048
#define MR_   (B_ * N_)          // 8192 rows for the token-major GEMMs

// GEMM tiling: 256 threads = 8 waves (wave32), 4(M) x 2(N) wave grid,
// each wave computes 32x32 via 2x2 WMMA fragments. Block tile 128x64, K-step 32.
#define TM 128
#define TN 64
#define TK 32
#define LDA_P 40                 // A LDS pitch (ushorts): 80B rows, 16B-aligned
#define LDB_P 72                 // B LDS pitch (ushorts): 144B rows, 16B-aligned

typedef __attribute__((ext_vector_type(16))) __bf16        v16bf;
typedef __attribute__((ext_vector_type(8)))  float         v8f;
typedef __attribute__((ext_vector_type(8)))  float         f32x8;
typedef __attribute__((ext_vector_type(4)))  unsigned int  u32x4;

union Frag  { v16bf v; u32x4 q[2]; unsigned short u[16]; };
union Pack8 { u32x4 v; unsigned short e[8]; };

__device__ __forceinline__ unsigned short f2bf(float f) {
  unsigned x = __float_as_uint(f);
  x += 0x7fffu + ((x >> 16) & 1u);      // round-to-nearest-even
  return (unsigned short)(x >> 16);
}

// ---------------------------------------------------------------------------
// Generic WMMA GEMM:  C[row, col] = sum_k A[row, k] * B'[col, k]   (+ epilogue)
//   A: bf16 row-major, lda (elements); MODE 5: A is fp32 (converted on stage)
//   B: bf16. Weight layout B[col][k] (N-major, k contiguous) by default;
//      MODE 5 uses K-major B[k][col] (col contiguous, ldb = per-k row stride).
//   LDS: As row-major [TM][TK]; Bs K-major [TK][TN] ; each A fragment is
//        two contiguous ds_load_b128 per lane; B fragments are 16 scalar LDS reads. All b128 addresses 16B-aligned.
//   Software pipeline: next tile's global loads are issued before the current
//   tile's WMMAs so load latency hides behind matrix math.
// Epilogue MODEs:
//   0: fp32 out, +bias                (attention output projection)
//   1: bf16 out, no bias              (q / k / v projections)
//   2: bf16 out, silu(x + bias)       (FFN up)
//   3: fp32 out, + bias + residual    (FFN down + final residual)
//   4: fp32 out, x*alpha + 1x1-conv u-bias  (attention logits)
//   5: bf16 out, A fp32, B K-major    (P @ V)
// ---------------------------------------------------------------------------
template<int MODE>
__global__ __launch_bounds__(256)
void gemm_k(const void* __restrict__ Av, const unsigned short* __restrict__ Bw,
            const float* __restrict__ bias, const float* __restrict__ resid,
            const float* __restrict__ u, const float* __restrict__ uW,
            const float* __restrict__ uB,
            float* __restrict__ outF, unsigned short* __restrict__ outH,
            int K, int lda, int ldb, int ldc, int ldr,
            long aZb, long aZh, long bZb, long bZh, long cZb, long cZh,
            float alpha) {
  const int bz = blockIdx.z;
  const int zb = bz / H_, zh = bz % H_;
  const long aOff = (long)zb * aZb + (long)zh * aZh;
  const long bOff = (long)zb * bZb + (long)zh * bZh;
  const long cOff = (long)zb * cZb + (long)zh * cZh;

  const int rowBlk = blockIdx.y * TM;
  const int colBlk = blockIdx.x * TN;

  __shared__ unsigned short As[TM][LDA_P];   // [row][k]
  __shared__ unsigned short Bs[TK][LDB_P];   // [k][n]  (K-major!)

  const int t    = threadIdx.x;
  const int w    = t >> 5;
  const int lane = t & 31;
  const int wm   = w >> 1;            // 0..3 -> M offset
  const int wn   = w & 1;             // 0..1 -> N offset
  const int half = lane >> 4;         // lane group
  const int mr   = lane & 15;

  const unsigned short* Ab = (const unsigned short*)Av;
  const float*          Af = (const float*)Av;

  // per-thread staging coordinates (chunk = 8 contiguous elements)
  const int ar0 = t >> 2,         ac0 = (t & 3) * 8;          // A chunk 0
  const int ar1 = (t + 256) >> 2, ac1 = ac0;                  // A chunk 1
  const int bkk5 = t >> 3, bnn5 = (t & 7) * 8;                // B (K-major src)
  const int bnn  = t >> 2, bkk  = (t & 3) * 8;                // B (N-major src)

  // register staging buffers (double-buffer vs. LDS)
  u32x4 aR0 = {}, aR1 = {}, bR = {};
  f32x8 aF0 = {}, aF1 = {};

  auto loadTile = [&](int k0) {
    if constexpr (MODE == 5) {
      aF0 = *(const f32x8*)&Af[aOff + (long)(rowBlk + ar0) * lda + (k0 + ac0)];
      aF1 = *(const f32x8*)&Af[aOff + (long)(rowBlk + ar1) * lda + (k0 + ac1)];
      bR  = *(const u32x4*)&Bw[bOff + (long)(k0 + bkk5) * ldb + (colBlk + bnn5)];
    } else {
      aR0 = *(const u32x4*)&Ab[aOff + (long)(rowBlk + ar0) * lda + (k0 + ac0)];
      aR1 = *(const u32x4*)&Ab[aOff + (long)(rowBlk + ar1) * lda + (k0 + ac1)];
      bR  = *(const u32x4*)&Bw[bOff + (long)(colBlk + bnn) * ldb + (k0 + bkk)];
    }
  };
  auto storeTile = [&]() {
    if constexpr (MODE == 5) {
      Pack8 p0, p1;
#pragma unroll
      for (int j = 0; j < 8; ++j) { p0.e[j] = f2bf(aF0[j]); p1.e[j] = f2bf(aF1[j]); }
      *(u32x4*)&As[ar0][ac0] = p0.v;
      *(u32x4*)&As[ar1][ac1] = p1.v;
      *(u32x4*)&Bs[bkk5][bnn5] = bR;            // both sides contiguous
    } else {
      *(u32x4*)&As[ar0][ac0] = aR0;
      *(u32x4*)&As[ar1][ac1] = aR1;
      Pack8 p; p.v = bR;                        // transpose into K-major LDS
#pragma unroll
      for (int j = 0; j < 8; ++j) Bs[bkk + j][bnn] = p.e[j];
    }
  };

  v8f acc[2][2];
  const v8f zf = {};
  acc[0][0] = zf; acc[0][1] = zf; acc[1][0] = zf; acc[1][1] = zf;

  auto compute = [&]() {
    Frag a[2], b[2];
#pragma unroll
    for (int fm = 0; fm < 2; ++fm) {
      const int rb = wm * 32 + fm * 16 + mr;        // A: lane -> M row
      a[fm].q[0] = *(const u32x4*)&As[rb][half * 8];        // k = half*8 + 0..7
      a[fm].q[1] = *(const u32x4*)&As[rb][16 + half * 8];   // k = 16+half*8+0..7
    }
#pragma unroll
    for (int fn = 0; fn < 2; ++fn) {
      const int nb = wn * 32 + fn * 16;             // B: lane -> N, elem -> K (table order)
#pragma unroll
      for (int s = 0; s < 16; ++s)
        b[fn].u[s] = Bs[(s < 8) ? (half * 8 + s) : (16 + half * 8 + (s - 8))][nb + mr];
    }
#pragma unroll
    for (int fm = 0; fm < 2; ++fm)
#pragma unroll
      for (int fn = 0; fn < 2; ++fn)
        acc[fm][fn] = __builtin_amdgcn_wmma_f32_16x16x32_bf16(
            false, a[fm].v, false, b[fn].v, (short)0, acc[fm][fn], false, false);
  };

  // ---- pipelined main loop ----
  loadTile(0);
  storeTile();
  __syncthreads();
  for (int k0 = TK; k0 < K; k0 += TK) {
    loadTile(k0);                // global loads in flight during compute
    if (k0 + TK < K) {           // prefetch tile after next (gfx1250 path)
      const int r = t >> 1, c = (t & 1) * 16;
      if constexpr (MODE == 5)
        __builtin_prefetch(&Af[aOff + (long)(rowBlk + r) * lda + (k0 + TK + c)], 0, 3);
      else
        __builtin_prefetch(&Ab[aOff + (long)(rowBlk + r) * lda + (k0 + TK + c)], 0, 3);
    }
    compute();                   // WMMAs hide the load latency
    __syncthreads();             // LDS consumed by all waves
    storeTile();                 // waits on loads land here, after compute
    __syncthreads();             // LDS ready
  }
  compute();

  // ---- epilogue: C element (vgpr i, lane) -> M = 8*half + i, N = mr ----
#pragma unroll
  for (int fm = 0; fm < 2; ++fm) {
#pragma unroll
    for (int fn = 0; fn < 2; ++fn) {
#pragma unroll
      for (int i = 0; i < 8; ++i) {
        const int row = rowBlk + wm * 32 + fm * 16 + half * 8 + i;
        const int col = colBlk + wn * 32 + fn * 16 + mr;
        const long cidx = cOff + (long)row * ldc + col;
        float v = acc[fm][fn][i];
        if constexpr (MODE == 0) {
          outF[cidx] = v + bias[col];
        } else if constexpr (MODE == 1) {
          outH[cidx] = f2bf(v);
        } else if constexpr (MODE == 2) {
          float s = v + bias[col];
          s = s / (1.f + __expf(-s));                  // silu
          outH[cidx] = f2bf(s);
        } else if constexpr (MODE == 3) {
          outF[cidx] = v + bias[col] + resid[(long)row * ldr + col];
        } else if constexpr (MODE == 4) {
          float ub = uB[zh];
#pragma unroll
          for (int c6 = 0; c6 < UE_; ++c6)
            ub += u[((long)(zb * UE_ + c6) * N_ + row) * N_ + col] * uW[zh * UE_ + c6];
          outF[cidx] = v * alpha + ub;
        } else if constexpr (MODE == 5) {
          outH[cidx] = f2bf(v);
        }
      }
    }
  }
}

// ---------------------------------------------------------------------------
// LayerNorm over D_=512 (256 thr x 2 elems), optional residual add,
// fp32 and/or bf16 output.
// ---------------------------------------------------------------------------
__global__ __launch_bounds__(256)
void ln_k(const float* __restrict__ in, const float* __restrict__ w,
          const float* __restrict__ b, const float* __restrict__ resid,
          float* __restrict__ outF, unsigned short* __restrict__ outH) {
  const long base = (long)blockIdx.x * D_;
  const int t = threadIdx.x;
  const float x0 = in[base + t], x1 = in[base + t + 256];
  __shared__ float red[256];
  red[t] = x0 + x1; __syncthreads();
  for (int s = 128; s > 0; s >>= 1) { if (t < s) red[t] += red[t + s]; __syncthreads(); }
  const float mu = red[0] * (1.f / D_);
  __syncthreads();
  const float d0 = x0 - mu, d1 = x1 - mu;
  red[t] = d0 * d0 + d1 * d1; __syncthreads();
  for (int s = 128; s > 0; s >>= 1) { if (t < s) red[t] += red[t + s]; __syncthreads(); }
  const float rstd = rsqrtf(red[0] * (1.f / D_) + 1e-5f);
  float y0 = d0 * rstd * w[t] + b[t];
  float y1 = d1 * rstd * w[t + 256] + b[t + 256];
  if (resid) { y0 += resid[base + t]; y1 += resid[base + t + 256]; }
  if (outF) { outF[base + t] = y0; outF[base + t + 256] = y1; }
  if (outH) { outH[base + t] = f2bf(y0); outH[base + t + 256] = f2bf(y1); }
}

// RMSNorm over D_=512 -> bf16
__global__ __launch_bounds__(256)
void rms_k(const float* __restrict__ in, const float* __restrict__ w,
           unsigned short* __restrict__ outH) {
  const long base = (long)blockIdx.x * D_;
  const int t = threadIdx.x;
  const float x0 = in[base + t], x1 = in[base + t + 256];
  __shared__ float red[256];
  red[t] = x0 * x0 + x1 * x1; __syncthreads();
  for (int s = 128; s > 0; s >>= 1) { if (t < s) red[t] += red[t + s]; __syncthreads(); }
  const float r = rsqrtf(red[0] * (1.f / D_) + 1.1920929e-07f);
  outH[base + t]       = f2bf(x0 * r * w[t]);
  outH[base + t + 256] = f2bf(x1 * r * w[t + 256]);
}

// In-place row softmax over N_=512 (one block per row), on d_out attn region
__global__ __launch_bounds__(256)
void softmax_k(float* __restrict__ p) {
  const long base = (long)blockIdx.x * N_;
  const int t = threadIdx.x;
  const float a = p[base + t], b = p[base + t + 256];
  __shared__ float red[256];
  red[t] = fmaxf(a, b); __syncthreads();
  for (int s = 128; s > 0; s >>= 1) { if (t < s) red[t] = fmaxf(red[t], red[t + s]); __syncthreads(); }
  const float m = red[0];
  __syncthreads();
  const float e0 = __expf(a - m), e1 = __expf(b - m);
  red[t] = e0 + e1; __syncthreads();
  for (int s = 128; s > 0; s >>= 1) { if (t < s) red[t] += red[t + s]; __syncthreads(); }
  const float inv = 1.f / red[0];
  p[base + t] = e0 * inv; p[base + t + 256] = e1 * inv;
}

// fp32 -> bf16 bulk convert (weights)
__global__ __launch_bounds__(256)
void cvt_k(const float* __restrict__ in, unsigned short* __restrict__ out, int n) {
  const int i = blockIdx.x * 256 + threadIdx.x;
  if (i < n) out[i] = f2bf(in[i]);
}

// ---------------------------------------------------------------------------
extern "C" void kernel_launch(void* const* d_in, const int* in_sizes, int n_in,
                              void* d_out, int out_size, void* d_ws, size_t ws_size,
                              hipStream_t stream) {
  const float* x     = (const float*)d_in[0];
  const float* u     = (const float*)d_in[1];
  const float* ln1_w = (const float*)d_in[2];
  const float* ln1_b = (const float*)d_in[3];
  const float* ln2_w = (const float*)d_in[4];
  const float* ln2_b = (const float*)d_in[5];
  const float* wq    = (const float*)d_in[6];
  const float* wk    = (const float*)d_in[7];
  const float* wv    = (const float*)d_in[8];
  const float* wo    = (const float*)d_in[9];
  const float* bo    = (const float*)d_in[10];
  const float* u_w   = (const float*)d_in[11];
  const float* u_b   = (const float*)d_in[12];
  const float* rms_w = (const float*)d_in[13];
  const float* w1    = (const float*)d_in[14];
  const float* b1    = (const float*)d_in[15];
  const float* w2    = (const float*)d_in[16];
  const float* b2    = (const float*)d_in[17];

  float* out1  = (float*)d_out;                                  // [B,N,D]
  float* attnW = (float*)d_out + (size_t)B_ * N_ * D_;           // [B,H,N,N]

  // workspace carve-up (all regions rewritten every call)
  char* ws = (char*)d_ws;
  size_t off = 0;
  auto carve = [&](size_t bytes) { void* p = ws + off; off = (off + bytes + 255) & ~(size_t)255; return p; };
  unsigned short* xn   = (unsigned short*)carve((size_t)MR_ * D_ * 2);
  unsigned short* qb   = (unsigned short*)carve((size_t)MR_ * D_ * 2);
  unsigned short* kb   = (unsigned short*)carve((size_t)MR_ * D_ * 2);
  unsigned short* vb   = (unsigned short*)carve((size_t)MR_ * D_ * 2);
  unsigned short* ctxb = (unsigned short*)carve((size_t)MR_ * D_ * 2);
  unsigned short* zb   = (unsigned short*)carve((size_t)MR_ * D_ * 2);
  unsigned short* zff  = (unsigned short*)carve((size_t)MR_ * DFF_ * 2);
  float*          aoF  = (float*)carve((size_t)MR_ * D_ * 4);    // attn_out (pre-LN2)
  float*          hF   = (float*)carve((size_t)MR_ * D_ * 4);    // h = LN2(attn_out)+x
  unsigned short* wqb  = (unsigned short*)carve((size_t)D_ * D_ * 2);
  unsigned short* wkb  = (unsigned short*)carve((size_t)D_ * D_ * 2);
  unsigned short* wvb  = (unsigned short*)carve((size_t)D_ * D_ * 2);
  unsigned short* wob  = (unsigned short*)carve((size_t)D_ * D_ * 2);
  unsigned short* w1b  = (unsigned short*)carve((size_t)DFF_ * D_ * 2);
  unsigned short* w2b  = (unsigned short*)carve((size_t)D_ * DFF_ * 2);
  (void)ws_size; (void)in_sizes; (void)n_in; (void)out_size;

  // 1) weights -> bf16
  cvt_k<<<dim3((D_ * D_) / 256), 256, 0, stream>>>(wq, wqb, D_ * D_);
  cvt_k<<<dim3((D_ * D_) / 256), 256, 0, stream>>>(wk, wkb, D_ * D_);
  cvt_k<<<dim3((D_ * D_) / 256), 256, 0, stream>>>(wv, wvb, D_ * D_);
  cvt_k<<<dim3((D_ * D_) / 256), 256, 0, stream>>>(wo, wob, D_ * D_);
  cvt_k<<<dim3((DFF_ * D_) / 256), 256, 0, stream>>>(w1, w1b, DFF_ * D_);
  cvt_k<<<dim3((D_ * DFF_) / 256), 256, 0, stream>>>(w2, w2b, D_ * DFF_);

  // 2) LN1: x -> xn (bf16)
  ln_k<<<dim3(MR_), 256, 0, stream>>>(x, ln1_w, ln1_b, nullptr, nullptr, xn);

  // 3) Q/K/V projections (MODE 1): [8192,512] = xn @ W^T, out bf16 in [B,N,D]
  const dim3 gProj(D_ / TN, MR_ / TM, 1);
  gemm_k<1><<<gProj, 256, 0, stream>>>(xn, wqb, nullptr, nullptr, nullptr, nullptr, nullptr,
                                       nullptr, qb, D_, D_, D_, D_, 0, 0,0,0,0,0,0, 1.f);
  gemm_k<1><<<gProj, 256, 0, stream>>>(xn, wkb, nullptr, nullptr, nullptr, nullptr, nullptr,
                                       nullptr, kb, D_, D_, D_, D_, 0, 0,0,0,0,0,0, 1.f);
  gemm_k<1><<<gProj, 256, 0, stream>>>(xn, wvb, nullptr, nullptr, nullptr, nullptr, nullptr,
                                       nullptr, vb, D_, D_, D_, D_, 0, 0,0,0,0,0,0, 1.f);

  // 4) logits (MODE 4): per (b,h)  S = (q_h k_h^T)*sqrt(HD) + ubias -> d_out
  {
    const dim3 g(N_ / TN, N_ / TM, B_ * H_);
    gemm_k<4><<<g, 256, 0, stream>>>(qb, kb, nullptr, nullptr, u, u_w, u_b,
                                     attnW, nullptr,
                                     HD_, D_, D_, N_, 0,
                                     (long)N_ * D_, (long)HD_,          // A (q)
                                     (long)N_ * D_, (long)HD_,          // B (k)
                                     (long)H_ * N_ * N_, (long)N_ * N_, // C
                                     8.0f /* sqrt(64) */);
  }

  // 5) softmax in place on the attn-weights output
  softmax_k<<<dim3(B_ * H_ * N_), 256, 0, stream>>>(attnW);

  // 6) ctx (MODE 5): per (b,h)  ctx = P @ v, P fp32 from d_out, out bf16 [B,N,D]
  {
    const dim3 g(HD_ / TN, N_ / TM, B_ * H_);
    gemm_k<5><<<g, 256, 0, stream>>>(attnW, vb, nullptr, nullptr, nullptr, nullptr, nullptr,
                                     nullptr, ctxb,
                                     N_, N_, D_, D_, 0,
                                     (long)H_ * N_ * N_, (long)N_ * N_, // A (P)
                                     (long)N_ * D_, (long)HD_,          // B (v), K-major
                                     (long)N_ * D_, (long)HD_,          // C (ctx)
                                     1.f);
  }

  // 7) output projection (MODE 0): attn_out = ctx @ wo^T + bo (fp32)
  gemm_k<0><<<gProj, 256, 0, stream>>>(ctxb, wob, bo, nullptr, nullptr, nullptr, nullptr,
                                       aoF, nullptr, D_, D_, D_, D_, 0, 0,0,0,0,0,0, 1.f);

  // 8) h = LN2(attn_out) + x   (fp32)
  ln_k<<<dim3(MR_), 256, 0, stream>>>(aoF, ln2_w, ln2_b, x, hF, nullptr);

  // 9) z = RMSNorm(h) (bf16)
  rms_k<<<dim3(MR_), 256, 0, stream>>>(hF, rms_w, zb);

  // 10) FFN up (MODE 2): zff = silu(z @ w1^T + b1) (bf16)
  {
    const dim3 g(DFF_ / TN, MR_ / TM, 1);
    gemm_k<2><<<g, 256, 0, stream>>>(zb, w1b, b1, nullptr, nullptr, nullptr, nullptr,
                                     nullptr, zff, D_, D_, D_, DFF_, 0, 0,0,0,0,0,0, 1.f);
  }

  // 11) FFN down + residual (MODE 3): out1 = zff @ w2^T + b2 + h (fp32)
  gemm_k<3><<<gProj, 256, 0, stream>>>(zff, w2b, b2, hF, nullptr, nullptr, nullptr,
                                       out1, nullptr, DFF_, DFF_, DFF_, D_, D_,
                                       0,0,0,0,0,0, 1.f);
}
